// ByteSeq2Seq_87625922773148
// MI455X (gfx1250) — matrix operand order repaired and checked by reading, not compiled, not run on hardware
//
#include <hip/hip_runtime.h>
#include <hip/hip_bf16.h>
#include <cstdint>

// ---------------------------------------------------------------------------
// ByteSeq2Seq on MI455X (gfx1250): persistent wave32 kernel, bf16 WMMA GEMMs,
// double-buffered async global->LDS weight staging (ASYNCcnt).
// V=256, E=128, H=512, B=128, S=1024, T=257
// ---------------------------------------------------------------------------

typedef __bf16 bf16_t;
typedef __attribute__((ext_vector_type(16))) __bf16 v16bf;
typedef __attribute__((ext_vector_type(8)))  __bf16 v8bf;
typedef __attribute__((ext_vector_type(8)))  float  v8f;

#define NBLK 16
#define NTHR 256
#define GRID_THREADS (NBLK * NTHR)

#define HB   512          // hidden
#define EB   128          // embed
#define GW   2048         // 4*H gate width
#define BATCH 128
#define SLEN 1024
#define DSTEPS 256        // T-1
#define VOC  256

// s_wait_asynccnt: builtin if available, else raw encoding via asm.
#if defined(__has_builtin)
#if __has_builtin(__builtin_amdgcn_s_wait_asynccnt)
#define WAIT_ASYNC(n) __builtin_amdgcn_s_wait_asynccnt(n)
#endif
#endif
#ifndef WAIT_ASYNC
#define WAIT_ASYNC(n) asm volatile("s_wait_asynccnt " #n ::: "memory")
#endif

struct Params {
    const int*    src;
    const int*    trg;
    const float*  enc_emb;
    const float*  dec_emb;
    const bf16_t *wih0e, *whh0e, *wih1e, *whh1e;
    const bf16_t *wih0d, *whh0d, *wih1d, *whh1d;
    const bf16_t *fcw;
    const float  *be0, *be1, *bd0, *bd1, *fcb;
    bf16_t *h0b, *h1b, *xb;
    float  *c0, *c1, *gbuf;
    int    *tokb;
    unsigned *bar;
    float  *out;
};

__device__ __forceinline__ float sigf(float x) { return 1.0f / (1.0f + __expf(-x)); }

// Device-wide barrier: all NBLK blocks arrive, spin on agent-scope counter.
// s_cluster_barrier is a NOP when not dispatched as a cluster.
__device__ __forceinline__ void gsync(unsigned* bar, unsigned gen) {
    __builtin_amdgcn_s_cluster_barrier();
    __syncthreads();
    if (threadIdx.x == 0) {
        __threadfence();
        __hip_atomic_fetch_add(bar, 1u, __ATOMIC_RELEASE, __HIP_MEMORY_SCOPE_AGENT);
        while (__hip_atomic_load(bar, __ATOMIC_ACQUIRE, __HIP_MEMORY_SCOPE_AGENT) < gen * NBLK) {
            __builtin_amdgcn_s_sleep(1);
        }
    }
    __syncthreads();
}

// A fragment (16x32 bf16, row-major):
// lane l=L&15 -> row m0+l ; hi=L>>4 selects contiguous K range hi*16..hi*16+15
__device__ __forceinline__ v16bf load_a_frag(const bf16_t* base, int row, int stride,
                                             int k0, int hi) {
    const bf16_t* p = base + (size_t)row * stride + k0 + hi * 16;
    union { v16bf v; v8bf h[2]; } u;
    u.h[0] = *(const v8bf*)(p);
    u.h[1] = *(const v8bf*)(p + 8);
    return u.v;
}

// B fragment (32x16 = K x N): lane l holds column n=l; lanes 0-15 K=k0..k0+15,
// lanes 16-31 K=k0+16..k0+31 (contiguous 16 values along K of row n of W).
__device__ __forceinline__ v16bf load_b_frag(const bf16_t* rowbase, int hi) {
    const bf16_t* p = rowbase + hi * 16;
    union { v16bf v; v8bf h[2]; } u;
    u.h[0] = *(const v8bf*)(p);
    u.h[1] = *(const v8bf*)(p + 8);
    return u.v;
}

// Asynchronously stage one 128x32 bf16 weight tile (rows n_base..+128, cols
// k0..k0+32 of W[*][K]) into an LDS buffer. Each of the 256 threads moves one
// 32-byte row segment with two global_load_async_to_lds_b128 (IOFFSET advances
// both the global and LDS addresses per the ISA pseudocode). ASYNCcnt += 2.
__device__ __forceinline__ void stage_async(const bf16_t* __restrict__ W, int K,
                                            int n_base, int k0, bf16_t* buf, int tid) {
    const int r  = tid >> 1;
    const int kh = (tid & 1) << 4;
    const bf16_t* s = W + (size_t)(n_base + r) * K + (k0 + kh);
    unsigned d = (unsigned)(uintptr_t)(buf + r * 32 + kh);
    asm volatile("global_load_async_to_lds_b128 %0, %1, off\n\t"
                 "global_load_async_to_lds_b128 %0, %1, off offset:16"
                 :: "v"(d), "v"(s)
                 : "memory");
}

// One LSTM gate pre-activation tile: g[128][n_base..n_base+128) =
//   bias + A1(128xK1) @ W1^T + A2(128x512) @ W2^T   (weights [2048][K] row-major)
__device__ __forceinline__ void gemm_block(const bf16_t* __restrict__ A1,
                                           const bf16_t* __restrict__ W1, int K1,
                                           const bf16_t* __restrict__ A2,
                                           const bf16_t* __restrict__ W2,
                                           const float*  __restrict__ bias,
                                           float* __restrict__ gout,
                                           bf16_t* lds) {
    const int tid  = threadIdx.x;
    const int wave = tid >> 5, lane = tid & 31, l = lane & 15, hi = lane >> 4;
    const int n_base = (int)(blockIdx.x << 7);   // 128-wide N slice
    const int m0     = wave << 4;                // 16-row M strip
    bf16_t* const buf0 = lds;
    bf16_t* const buf1 = lds + 128 * 32;

    v8f acc[8];
#pragma unroll
    for (int nt = 0; nt < 8; ++nt) {
        float bv = bias[n_base + nt * 16 + l];
#pragma unroll
        for (int r = 0; r < 8; ++r) acc[nt][r] = bv;
    }

    for (int pass = 0; pass < 2; ++pass) {
        const bf16_t* A = pass ? A2 : A1;
        const bf16_t* W = pass ? W2 : W1;
        const int K = pass ? HB : K1;
        // Prologue: fill buffer 0 with the first K-tile.
        stage_async(W, K, n_base, 0, buf0, tid);
        for (int k0 = 0; k0 < K; k0 += 32) {
            bf16_t* const cur = ((k0 >> 5) & 1) ? buf1 : buf0;
            bf16_t* const nxt = ((k0 >> 5) & 1) ? buf0 : buf1;
            if (k0 + 32 < K) {
                // Overlap next tile's fill with this tile's WMMAs. Overwrite of
                // nxt is safe: its consumers passed the post-consume barrier of
                // the previous iteration.
                stage_async(W, K, n_base, k0 + 32, nxt, tid);
                WAIT_ASYNC(2);   // drain this tile's pair; next tile in flight
            } else {
                WAIT_ASYNC(0);
            }
            __syncthreads();      // tile 'cur' now complete and visible
            v16bf a = load_a_frag(A, m0 + l, K, k0, hi);
#pragma unroll
            for (int nt = 0; nt < 8; ++nt) {
                v16bf b = load_b_frag(cur + (nt * 16 + l) * 32, hi);
                acc[nt] = __builtin_amdgcn_wmma_f32_16x16x32_bf16(
                    false, a, false, b, (short)0, acc[nt], false, false);
            }
            __syncthreads();      // all waves done with 'cur'
        }
    }

    // D layout: element (m0 + hi*8 + r, n_base + nt*16 + l) = acc[nt][r]
#pragma unroll
    for (int nt = 0; nt < 8; ++nt) {
        int n = n_base + nt * 16 + l;
#pragma unroll
        for (int r = 0; r < 8; ++r) {
            int m = m0 + (hi << 3) + r;
            gout[m * GW + n] = acc[nt][r];
        }
    }
}

// Elementwise LSTM gates over 128x512; updates c (f32) and h (bf16).
__device__ __forceinline__ void gates_block(const float* __restrict__ g,
                                            float* __restrict__ c,
                                            bf16_t* __restrict__ h) {
    const int gt = (int)(blockIdx.x * blockDim.x + threadIdx.x);
    for (int idx = gt; idx < BATCH * HB; idx += GRID_THREADS) {
        const float* gr = g + ((idx >> 9) << 11);   // row base, stride 2048
        const int j = idx & (HB - 1);
        float iv = gr[j], fv = gr[j + HB], gv = gr[j + 2 * HB], ov = gr[j + 3 * HB];
        float cn = sigf(fv) * c[idx] + sigf(iv) * tanhf(gv);
        c[idx] = cn;
        h[idx] = (bf16_t)(sigf(ov) * tanhf(cn));
    }
}

// logits[128][256] = h1 @ fc_w^T + fc_b ; each block owns a 16-wide N slice.
__device__ __forceinline__ void fc_block(const bf16_t* __restrict__ h1b,
                                         const bf16_t* __restrict__ fcw,
                                         const float*  __restrict__ fcb,
                                         float* __restrict__ out, int t) {
    const int tid  = threadIdx.x;
    const int wave = tid >> 5, lane = tid & 31, l = lane & 15, hi = lane >> 4;
    const int nb   = (int)(blockIdx.x << 4);
    const int m0   = wave << 4;
    v8f acc;
    float bv = fcb[nb + l];
#pragma unroll
    for (int r = 0; r < 8; ++r) acc[r] = bv;
    const bf16_t* rowb = fcw + (size_t)(nb + l) * HB;
    for (int k0 = 0; k0 < HB; k0 += 32) {
        v16bf a = load_a_frag(h1b, m0 + l, HB, k0, hi);
        v16bf b = load_b_frag(rowb + k0, hi);
        if (k0 + 32 < HB) __builtin_prefetch(rowb + k0 + 32, 0, 1);
        acc = __builtin_amdgcn_wmma_f32_16x16x32_bf16(
            false, a, false, b, (short)0, acc, false, false);
    }
#pragma unroll
    for (int r = 0; r < 8; ++r) {
        int m = m0 + (hi << 3) + r;
        out[(size_t)m * (DSTEPS * VOC) + (size_t)t * VOC + nb + l] = acc[r];
    }
}

// First-occurrence argmax over 256 logits per row (jnp.argmax semantics).
__device__ __forceinline__ void argmax_block(const float* __restrict__ out, int t,
                                             int* __restrict__ tokb) {
    const int tid  = threadIdx.x;
    const int wave = tid >> 5, lane = tid & 31;
    const int b = (int)(blockIdx.x * 8 + wave);   // 16 blocks * 8 waves = 128 rows
    const float* row = out + (size_t)b * (DSTEPS * VOC) + (size_t)t * VOC;
    float best = -3.402823466e+38f;
    int   bi   = 0;
    for (int v = lane; v < VOC; v += 32) {
        float x = row[v];
        if (x > best) { best = x; bi = v; }
    }
    for (int off = 16; off >= 1; off >>= 1) {
        float ob = __shfl_xor(best, off, 32);
        int   oi = __shfl_xor(bi,   off, 32);
        if (ob > best || (ob == best && oi < bi)) { best = ob; bi = oi; }
    }
    if (lane == 0) tokb[b] = bi;
}

__global__ __launch_bounds__(NTHR) void k_seq2seq(Params p) {
    __shared__ __align__(16) bf16_t ldsW[2 * 128 * 32];   // double-buffered tile
    const int gt = (int)(blockIdx.x * blockDim.x + threadIdx.x);
    unsigned gen = 0;

    // ---- init recurrent state ----
    for (int idx = gt; idx < BATCH * HB; idx += GRID_THREADS) {
        p.c0[idx] = 0.0f; p.c1[idx] = 0.0f;
        p.h0b[idx] = (bf16_t)0.0f; p.h1b[idx] = (bf16_t)0.0f;
    }
    gsync(p.bar, ++gen);

    // ---- encoder: 1024 steps, both layers fused per step ----
    for (int t = 0; t < SLEN; ++t) {
        for (int idx = gt; idx < BATCH * EB; idx += GRID_THREADS) {
            int b = idx >> 7, e = idx & (EB - 1);
            p.xb[idx] = (bf16_t)p.enc_emb[(size_t)p.src[b * SLEN + t] * EB + e];
        }
        gsync(p.bar, ++gen);
        gemm_block(p.xb, p.wih0e, EB, p.h0b, p.whh0e, p.be0, p.gbuf, ldsW);
        gsync(p.bar, ++gen);
        gates_block(p.gbuf, p.c0, p.h0b);
        gsync(p.bar, ++gen);
        gemm_block(p.h0b, p.wih1e, HB, p.h1b, p.whh1e, p.be1, p.gbuf, ldsW);
        gsync(p.bar, ++gen);
        gates_block(p.gbuf, p.c1, p.h1b);
        gsync(p.bar, ++gen);
    }

    // ---- decoder init: tok = trg[:, 0] ----
    for (int idx = gt; idx < BATCH; idx += GRID_THREADS)
        p.tokb[idx] = p.trg[idx * (DSTEPS + 1)];
    gsync(p.bar, ++gen);

    // ---- greedy decode: 256 steps ----
    for (int t = 0; t < DSTEPS; ++t) {
        for (int idx = gt; idx < BATCH * EB; idx += GRID_THREADS) {
            int b = idx >> 7, e = idx & (EB - 1);
            p.xb[idx] = (bf16_t)p.dec_emb[(size_t)p.tokb[b] * EB + e];
        }
        gsync(p.bar, ++gen);
        gemm_block(p.xb, p.wih0d, EB, p.h0b, p.whh0d, p.bd0, p.gbuf, ldsW);
        gsync(p.bar, ++gen);
        gates_block(p.gbuf, p.c0, p.h0b);
        gsync(p.bar, ++gen);
        gemm_block(p.h0b, p.wih1d, HB, p.h1b, p.whh1d, p.bd1, p.gbuf, ldsW);
        gsync(p.bar, ++gen);
        gates_block(p.gbuf, p.c1, p.h1b);
        gsync(p.bar, ++gen);
        fc_block(p.h1b, p.fcw, p.fcb, p.out, t);
        gsync(p.bar, ++gen);
        argmax_block(p.out, t, p.tokb);
        gsync(p.bar, ++gen);
    }
}

// ---------------------------- prep kernels ---------------------------------
__global__ void k_cvt_bf16(const float* __restrict__ s, bf16_t* __restrict__ d, int n) {
    int i = (int)(blockIdx.x * blockDim.x + threadIdx.x);
    if (i < n) d[i] = (bf16_t)s[i];
}
__global__ void k_bias_sum(const float* __restrict__ a, const float* __restrict__ b,
                           float* __restrict__ d, int n) {
    int i = (int)(blockIdx.x * blockDim.x + threadIdx.x);
    if (i < n) d[i] = a[i] + b[i];
}
__global__ void k_copy_f32(const float* __restrict__ a, float* __restrict__ d, int n) {
    int i = (int)(blockIdx.x * blockDim.x + threadIdx.x);
    if (i < n) d[i] = a[i];
}

// ------------------------------- host --------------------------------------
extern "C" void kernel_launch(void* const* d_in, const int* in_sizes, int n_in,
                              void* d_out, int out_size, void* d_ws, size_t ws_size,
                              hipStream_t stream) {
    (void)in_sizes; (void)n_in; (void)out_size; (void)ws_size;
    const int*   src     = (const int*)  d_in[0];
    const int*   trg     = (const int*)  d_in[1];
    const float* enc_emb = (const float*)d_in[2];
    const float* ew_ih0  = (const float*)d_in[3];
    const float* ew_hh0  = (const float*)d_in[4];
    const float* eb_ih0  = (const float*)d_in[5];
    const float* eb_hh0  = (const float*)d_in[6];
    const float* ew_ih1  = (const float*)d_in[7];
    const float* ew_hh1  = (const float*)d_in[8];
    const float* eb_ih1  = (const float*)d_in[9];
    const float* eb_hh1  = (const float*)d_in[10];
    const float* dec_emb = (const float*)d_in[11];
    const float* dw_ih0  = (const float*)d_in[12];
    const float* dw_hh0  = (const float*)d_in[13];
    const float* db_ih0  = (const float*)d_in[14];
    const float* db_hh0  = (const float*)d_in[15];
    const float* dw_ih1  = (const float*)d_in[16];
    const float* dw_hh1  = (const float*)d_in[17];
    const float* db_ih1  = (const float*)d_in[18];
    const float* db_hh1  = (const float*)d_in[19];
    const float* fc_w    = (const float*)d_in[20];
    const float* fc_b    = (const float*)d_in[21];

    char* ws = (char*)d_ws;
    size_t off = 0;
    auto take = [&](size_t bytes) -> char* {
        char* p = ws + off;
        off += (bytes + 255) & ~(size_t)255;
        return p;
    };

    unsigned* bar = (unsigned*)take(256);
    bf16_t* wih0e = (bf16_t*)take((size_t)GW * EB * 2);
    bf16_t* whh0e = (bf16_t*)take((size_t)GW * HB * 2);
    bf16_t* wih1e = (bf16_t*)take((size_t)GW * HB * 2);
    bf16_t* whh1e = (bf16_t*)take((size_t)GW * HB * 2);
    bf16_t* wih0d = (bf16_t*)take((size_t)GW * EB * 2);
    bf16_t* whh0d = (bf16_t*)take((size_t)GW * HB * 2);
    bf16_t* wih1d = (bf16_t*)take((size_t)GW * HB * 2);
    bf16_t* whh1d = (bf16_t*)take((size_t)GW * HB * 2);
    bf16_t* fcw   = (bf16_t*)take((size_t)VOC * HB * 2);
    float*  be0   = (float*) take(GW * 4);
    float*  be1   = (float*) take(GW * 4);
    float*  bd0   = (float*) take(GW * 4);
    float*  bd1   = (float*) take(GW * 4);
    float*  fcb   = (float*) take(VOC * 4);
    bf16_t* h0b   = (bf16_t*)take((size_t)BATCH * HB * 2);
    bf16_t* h1b   = (bf16_t*)take((size_t)BATCH * HB * 2);
    bf16_t* xb    = (bf16_t*)take((size_t)BATCH * EB * 2);
    float*  c0    = (float*) take((size_t)BATCH * HB * 4);
    float*  c1    = (float*) take((size_t)BATCH * HB * 4);
    float*  gbuf  = (float*) take((size_t)BATCH * GW * 4);
    int*    tokb  = (int*)   take(BATCH * 4);

    hipMemsetAsync(bar, 0, 256, stream);

    auto cvt = [&](const float* s, bf16_t* d, int n) {
        k_cvt_bf16<<<dim3((n + 255) / 256), dim3(256), 0, stream>>>(s, d, n);
    };
    cvt(ew_ih0, wih0e, GW * EB);
    cvt(ew_hh0, whh0e, GW * HB);
    cvt(ew_ih1, wih1e, GW * HB);
    cvt(ew_hh1, whh1e, GW * HB);
    cvt(dw_ih0, wih0d, GW * EB);
    cvt(dw_hh0, whh0d, GW * HB);
    cvt(dw_ih1, wih1d, GW * HB);
    cvt(dw_hh1, whh1d, GW * HB);
    cvt(fc_w,   fcw,   VOC * HB);

    k_bias_sum<<<dim3((GW + 255) / 256), dim3(256), 0, stream>>>(eb_ih0, eb_hh0, be0, GW);
    k_bias_sum<<<dim3((GW + 255) / 256), dim3(256), 0, stream>>>(eb_ih1, eb_hh1, be1, GW);
    k_bias_sum<<<dim3((GW + 255) / 256), dim3(256), 0, stream>>>(db_ih0, db_hh0, bd0, GW);
    k_bias_sum<<<dim3((GW + 255) / 256), dim3(256), 0, stream>>>(db_ih1, db_hh1, bd1, GW);
    k_copy_f32<<<dim3(1), dim3(256), 0, stream>>>(fc_b, fcb, VOC);

    Params p;
    p.src = src; p.trg = trg; p.enc_emb = enc_emb; p.dec_emb = dec_emb;
    p.wih0e = wih0e; p.whh0e = whh0e; p.wih1e = wih1e; p.whh1e = whh1e;
    p.wih0d = wih0d; p.whh0d = whh0d; p.wih1d = wih1d; p.whh1d = whh1d;
    p.fcw = fcw; p.be0 = be0; p.be1 = be1; p.bd0 = bd0; p.bd1 = bd1; p.fcb = fcb;
    p.h0b = h0b; p.h1b = h1b; p.xb = xb; p.c0 = c0; p.c1 = c1; p.gbuf = gbuf;
    p.tokb = tokb; p.bar = bar; p.out = (float*)d_out;

    k_seq2seq<<<dim3(NBLK), dim3(NTHR), 0, stream>>>(p);
}
